// BilinearCrossAttention_3264175145728
// MI455X (gfx1250) — matrix operand order repaired and checked by reading, not compiled, not run on hardware
//
#include <hip/hip_runtime.h>

typedef __attribute__((ext_vector_type(16))) _Float16 v16h;
typedef __attribute__((ext_vector_type(8)))  _Float16 v8h;
typedef __attribute__((ext_vector_type(8)))  float    v8f;

#define D_MODEL   1024
#define NUM_HEADS 16
#define D_K       64
#define RANK      16
#define BATCH     2
#define SEQ       2048

static __device__ __forceinline__ v8f wmma_f16(v16h a, v16h b, v8f c) {
  // (neg_a, A, neg_b, B, c_mod, C, reuse_a, reuse_b)
  return __builtin_amdgcn_wmma_f32_16x16x32_f16(false, a, false, b, (short)0, c,
                                                false, false);
}

// ---------------------------------------------------------------------------
// f32 -> f16 conversion
// ---------------------------------------------------------------------------
__global__ void cvt_f32_f16(const float* __restrict__ in,
                            _Float16* __restrict__ out, int n) {
  int i = blockIdx.x * blockDim.x + threadIdx.x;
  if (i < n) out[i] = (_Float16)in[i];
}

// ---------------------------------------------------------------------------
// Y[m,n] = sum_k A[m,k] * W[n,k] + bias[n]    (torch Linear: y = x @ W.T + b)
// A: [M,K] f16, W: [N,K] f16 (row-major), Y: [M,N] f32.
// One wave computes a 32x64 output tile (2 A-frags x 4 B-frags = 8 WMMAs per
// 32-wide K step; B fragments reused x2 from registers). A and B fragments use
// the same contiguous K order, k = 16*khalf + s (wave32).
// ---------------------------------------------------------------------------
__global__ void __launch_bounds__(256) gemm_xwT(
    const _Float16* __restrict__ A, const _Float16* __restrict__ W,
    const float* __restrict__ bias, float* __restrict__ Y,
    int M, int N, int K) {
  const int lane   = threadIdx.x & 31;
  const int wave   = blockIdx.x * (blockDim.x >> 5) + (threadIdx.x >> 5);
  const int tilesN = N >> 6;
  const int mTile  = wave / tilesN;          // 32-row tile
  const int nTile  = wave - mTile * tilesN;  // 64-col tile
  if (mTile * 32 >= M) return;
  const int col   = lane & 15;   // A row-in-tile / B column-in-tile
  const int khalf = lane >> 4;   // which K-half of the fragment this lane holds
  const int row0  = mTile * 32 + col;
  const int row1  = row0 + 16;
  const int n0    = nTile * 64;
  const _Float16* arow0 = A + (size_t)row0 * K;
  const _Float16* arow1 = A + (size_t)row1 * K;

  v8f acc[8] = {};
  for (int k0 = 0; k0 < K; k0 += 32) {
    // A 16x32: lane holds M=col; K = 16*khalf..+15 contiguous
    v8h a0lo = *(const v8h*)(arow0 + k0 + khalf * 16);
    v8h a0hi = *(const v8h*)(arow0 + k0 + khalf * 16 + 8);
    v8h a1lo = *(const v8h*)(arow1 + k0 + khalf * 16);
    v8h a1hi = *(const v8h*)(arow1 + k0 + khalf * 16 + 8);
    v16h a0, a1;
#pragma unroll
    for (int i = 0; i < 8; ++i) {
      a0[i] = a0lo[i]; a0[8 + i] = a0hi[i];
      a1[i] = a1lo[i]; a1[8 + i] = a1hi[i];
    }
#pragma unroll
    for (int t = 0; t < 4; ++t) {
      // B 32x16: lane holds N=col, K = 16*khalf..+15 contiguous in W row n
      const int n = n0 + t * 16 + col;
      v16h b = *(const v16h*)(W + (size_t)n * K + k0 + khalf * 16);
      acc[t]     = wmma_f16(a0, b, acc[t]);
      acc[4 + t] = wmma_f16(a1, b, acc[4 + t]);
    }
  }
#pragma unroll
  for (int t = 0; t < 4; ++t) {
    const int n  = n0 + t * 16 + col;
    const float bv = bias[n];
#pragma unroll
    for (int i = 0; i < 8; ++i) {
      // D-frag: VGPR i => row i + 8*khalf
      const int m0 = mTile * 32 + i + 8 * khalf;
      Y[(size_t)m0 * N + n]        = acc[t][i] + bv;
      Y[(size_t)(m0 + 16) * N + n] = acc[4 + t][i] + bv;
    }
  }
}

// ---------------------------------------------------------------------------
// Qp[bh,l,r] = sum_d Q[(b*L+l), h*64+d] * U[h,d,r]    -> f16 [B*H, L, 16]
// ---------------------------------------------------------------------------
__global__ void rank_proj(const float* __restrict__ Qf,  // [B*L, D_MODEL]
                          const float* __restrict__ Um,  // [H, 64, 16]
                          _Float16* __restrict__ Outp) { // [B*H, L, 16]
  const int tid = blockIdx.x * blockDim.x + threadIdx.x;
  const int r  = tid & 15;
  const int l  = (tid >> 4) & (SEQ - 1);
  const int bh = tid >> 4 >> 11;           // tid / (16*SEQ)
  const int b  = bh / NUM_HEADS;
  const int h  = bh % NUM_HEADS;
  const float* q = Qf + ((size_t)(b * SEQ + l)) * D_MODEL + h * D_K;
  const float* u = Um + ((size_t)h * D_K) * RANK + r;
  float acc = 0.f;
#pragma unroll 8
  for (int d = 0; d < D_K; ++d) acc += q[d] * u[(size_t)d * RANK];
  Outp[tid] = (_Float16)acc;
}

// ---------------------------------------------------------------------------
// Vt[bh, d, l] = (f16) V[(b*L+l), h*64+d]   -> f16 [B*H, 64, L]
// ---------------------------------------------------------------------------
__global__ void transpose_v(const float* __restrict__ Vf,
                            _Float16* __restrict__ Vt) {
  const int tid = blockIdx.x * blockDim.x + threadIdx.x;  // B*H*64*L
  const int l  = tid & (SEQ - 1);
  const int d  = (tid >> 11) & (D_K - 1);
  const int bh = tid >> 17;                 // / (64*2048)
  const int b  = bh / NUM_HEADS;
  const int h  = bh % NUM_HEADS;
  Vt[tid] = (_Float16)Vf[((size_t)(b * SEQ + l)) * D_MODEL + h * D_K + d];
}

// ---------------------------------------------------------------------------
// Flash attention: per wave, one 16-query tile of one (b,h).
//   S = Qp * Kp^T / sqrt(RANK)  (rank padded 16->32 with zeros)
//   online softmax over 32-key tiles, P bounced via LDS into A-fragment,
//   ctx16[q, h*64+d] accumulated with 4 WMMAs per key tile.
// ---------------------------------------------------------------------------
__global__ void __launch_bounds__(256) flash_attn(
    const _Float16* __restrict__ Qp,   // [B*H, L, 16]
    const _Float16* __restrict__ Kp,   // [B*H, L, 16]
    const _Float16* __restrict__ Vt,   // [B*H, 64, L]
    const unsigned char* __restrict__ mask,  // [B, L]
    _Float16* __restrict__ ctx) {      // [B*L, D_MODEL] f16
  __shared__ __align__(32) _Float16 Pl[8][16][32];  // per-wave P staging
  const int lane   = threadIdx.x & 31;
  const int wv     = threadIdx.x >> 5;
  const int gwave  = blockIdx.x * 8 + wv;
  const int qTiles = SEQ >> 4;
  const int bh = gwave / qTiles;
  const int qt = gwave % qTiles;
  const int b  = bh / NUM_HEADS;
  const int h  = bh % NUM_HEADS;
  const int col   = lane & 15;
  const int khalf = lane >> 4;
  const int qrow  = qt * 16 + col;

  // Qp A-fragment: rank 0..15 real, 16..31 zero-padded
  v16h aq = {};
  if (khalf == 0) {
    aq = *(const v16h*)(Qp + ((size_t)bh * SEQ + qrow) * RANK);
  }

  v8f o[4] = {};
  v8f mrow, srow;
#pragma unroll
  for (int i = 0; i < 8; ++i) { mrow[i] = -1e30f; srow[i] = 0.f; }

  for (int k0 = 0; k0 < SEQ; k0 += 32) {
    v8f s[2];
#pragma unroll
    for (int kt = 0; kt < 2; ++kt) {
      // Kp^T B-fragment: lanes 0-15 hold kk(rank)=0..15 = full Kp row (32B);
      // lanes 16-31 are the zero pad (rank 16..31).
      v16h bk = {};
      if (khalf == 0) {
        const int key = k0 + kt * 16 + col;
        bk = *(const v16h*)(Kp + ((size_t)bh * SEQ + key) * RANK);
      }
      v8f z = {};
      v8f sv = wmma_f16(aq, bk, z);
      const int keyc = k0 + kt * 16 + col;            // D-frag column = lane&15
      const bool msk = mask[(size_t)b * SEQ + keyc] != 0;
#pragma unroll
      for (int i = 0; i < 8; ++i)
        sv[i] = msk ? -10000.f : sv[i] * 0.25f;       // 1/sqrt(16)
      s[kt] = sv;
    }
    // Online softmax; row i+8*khalf statistics via 16-lane xor reductions.
#pragma unroll
    for (int i = 0; i < 8; ++i) {
      float t = fmaxf(s[0][i], s[1][i]);
      t = fmaxf(t, __shfl_xor(t, 1));
      t = fmaxf(t, __shfl_xor(t, 2));
      t = fmaxf(t, __shfl_xor(t, 4));
      t = fmaxf(t, __shfl_xor(t, 8));
      const float mn   = fmaxf(mrow[i], t);
      const float corr = __expf(mrow[i] - mn);
      mrow[i] = mn;
      const float p0 = __expf(s[0][i] - mn);
      const float p1 = __expf(s[1][i] - mn);
      float ps = p0 + p1;
      ps += __shfl_xor(ps, 1);
      ps += __shfl_xor(ps, 2);
      ps += __shfl_xor(ps, 4);
      ps += __shfl_xor(ps, 8);
      srow[i] = srow[i] * corr + ps;
      o[0][i] *= corr; o[1][i] *= corr; o[2][i] *= corr; o[3][i] *= corr;
      const int prow = i + 8 * khalf;
      Pl[wv][prow][col]      = (_Float16)p0;
      Pl[wv][prow][col + 16] = (_Float16)p1;
    }
    // P A-fragment (16x32) from LDS (same-wave DS ops are ordered)
    v16h ap;
    {
      v8h plo = *(const v8h*)&Pl[wv][col][khalf * 16];
      v8h phi = *(const v8h*)&Pl[wv][col][khalf * 16 + 8];
#pragma unroll
      for (int i = 0; i < 8; ++i) { ap[i] = plo[i]; ap[8 + i] = phi[i]; }
    }
    // P (16x32) x V (32x64): B-frag is 32B contiguous along L in Vt row d
#pragma unroll
    for (int t4 = 0; t4 < 4; ++t4) {
      const int d = t4 * 16 + col;
      v16h bv = *(const v16h*)(Vt + ((size_t)bh * D_K + d) * SEQ + k0 + khalf * 16);
      o[t4] = wmma_f16(ap, bv, o[t4]);
    }
  }
#pragma unroll
  for (int t4 = 0; t4 < 4; ++t4) {
    const int d = t4 * 16 + col;
#pragma unroll
    for (int i = 0; i < 8; ++i) {
      const int q = qt * 16 + i + 8 * khalf;
      ctx[((size_t)b * SEQ + q) * D_MODEL + h * D_K + d] =
          (_Float16)(o[t4][i] / srow[i]);
    }
  }
}

// ---------------------------------------------------------------------------
extern "C" void kernel_launch(void* const* d_in, const int* in_sizes, int n_in,
                              void* d_out, int out_size, void* d_ws, size_t ws_size,
                              hipStream_t stream) {
  (void)in_sizes; (void)n_in; (void)out_size; (void)ws_size;
  const float* x_q  = (const float*)d_in[0];
  const float* x_kv = (const float*)d_in[1];
  const float* Wq   = (const float*)d_in[2];
  const float* bq   = (const float*)d_in[3];
  const float* Wk   = (const float*)d_in[4];
  const float* bk   = (const float*)d_in[5];
  const float* Wv   = (const float*)d_in[6];
  const float* bv   = (const float*)d_in[7];
  const float* Wo   = (const float*)d_in[8];
  const float* bo   = (const float*)d_in[9];
  const float* U    = (const float*)d_in[10];
  const float* Vb   = (const float*)d_in[11];
  const unsigned char* pmask = (const unsigned char*)d_in[12];
  float* out = (float*)d_out;

  const int ML = BATCH * SEQ;  // 4096
  char* ws = (char*)d_ws;
  size_t off = 0;
  auto alloc = [&](size_t bytes) -> void* {
    void* p = ws + off;
    off += (bytes + 255) & ~(size_t)255;
    return p;
  };
  _Float16* xq16  = (_Float16*)alloc((size_t)ML * D_MODEL * 2);
  _Float16* xkv16 = (_Float16*)alloc((size_t)ML * D_MODEL * 2);
  _Float16* wq16  = (_Float16*)alloc((size_t)D_MODEL * D_MODEL * 2);
  _Float16* wk16  = (_Float16*)alloc((size_t)D_MODEL * D_MODEL * 2);
  _Float16* wv16  = (_Float16*)alloc((size_t)D_MODEL * D_MODEL * 2);
  _Float16* wo16  = (_Float16*)alloc((size_t)D_MODEL * D_MODEL * 2);
  float*    Qf    = (float*)alloc((size_t)ML * D_MODEL * 4);
  float*    Kf    = (float*)alloc((size_t)ML * D_MODEL * 4);
  float*    Vf    = (float*)alloc((size_t)ML * D_MODEL * 4);
  _Float16* Qp    = (_Float16*)alloc((size_t)BATCH * NUM_HEADS * SEQ * RANK * 2);
  _Float16* Kp    = (_Float16*)alloc((size_t)BATCH * NUM_HEADS * SEQ * RANK * 2);
  _Float16* Vt    = (_Float16*)alloc((size_t)BATCH * NUM_HEADS * D_K * SEQ * 2);
  _Float16* ctx16 = (_Float16*)alloc((size_t)ML * D_MODEL * 2);

  const int nX = ML * D_MODEL;          // 4,194,304
  const int nW = D_MODEL * D_MODEL;     // 1,048,576
  cvt_f32_f16<<<(nX + 255) / 256, 256, 0, stream>>>(x_q,  xq16,  nX);
  cvt_f32_f16<<<(nX + 255) / 256, 256, 0, stream>>>(x_kv, xkv16, nX);
  cvt_f32_f16<<<(nW + 255) / 256, 256, 0, stream>>>(Wq, wq16, nW);
  cvt_f32_f16<<<(nW + 255) / 256, 256, 0, stream>>>(Wk, wk16, nW);
  cvt_f32_f16<<<(nW + 255) / 256, 256, 0, stream>>>(Wv, wv16, nW);
  cvt_f32_f16<<<(nW + 255) / 256, 256, 0, stream>>>(Wo, wo16, nW);

  // (M/32)*(N/64) = 128*16 = 2048 waves = 256 blocks of 8 waves
  gemm_xwT<<<256, 256, 0, stream>>>(xq16,  wq16, bq, Qf, ML, D_MODEL, D_MODEL);
  gemm_xwT<<<256, 256, 0, stream>>>(xkv16, wk16, bk, Kf, ML, D_MODEL, D_MODEL);
  gemm_xwT<<<256, 256, 0, stream>>>(xkv16, wv16, bv, Vf, ML, D_MODEL, D_MODEL);

  const int nP = BATCH * NUM_HEADS * SEQ * RANK;  // 1,048,576
  rank_proj<<<nP / 256, 256, 0, stream>>>(Qf, U,  Qp);
  rank_proj<<<nP / 256, 256, 0, stream>>>(Kf, Vb, Kp);

  const int nV = BATCH * NUM_HEADS * D_K * SEQ;   // 4,194,304
  transpose_v<<<nV / 256, 256, 0, stream>>>(Vf, Vt);

  // B*H*(L/16) = 4096 waves = 512 blocks
  flash_attn<<<512, 256, 0, stream>>>(Qp, Kp, Vt, pmask, ctx16);

  gemm_xwT<<<256, 256, 0, stream>>>(ctx16, wo16, bo, out, ML, D_MODEL, D_MODEL);
}
